// FrequencySpatialAttention_61942018343486
// MI455X (gfx1250) — matrix operand order repaired and checked by reading, not compiled, not run on hardware
//
#include <hip/hip_runtime.h>
#include <hip/hip_bf16.h>

typedef __attribute__((ext_vector_type(16))) _Float16 v16h;
typedef __attribute__((ext_vector_type(8)))  float    v8f;

#define PI_F 3.14159265358979323846f

__device__ __forceinline__ float dct1(float p, float uf, float scale) {
  // cos(pi*(p+0.5)*u/8) * scale
  return __cosf(PI_F * (p + 0.5f) * uf * 0.125f) * scale;
}

// ---------------------------------------------------------------------------
// Kernel 1: x_mean[b,h,w] = mean_c x[b,c,h,w]    (streams 268 MB, writes 4 MB)
// one thread = 4 consecutive (h,w) positions, float4 loads, 64-deep C loop
// ---------------------------------------------------------------------------
__global__ __launch_bounds__(256) void k_chan_mean(const float* __restrict__ x,
                                                   float* __restrict__ xmean) {
  const int tid = blockIdx.x * 256 + threadIdx.x;      // 262144 threads total
  const int b   = tid >> 14;                           // 16384 float4 per batch
  const int hw4 = tid & 16383;
  const float4* px = reinterpret_cast<const float4*>(x) + (size_t)b * (64u * 16384u) + hw4;
  float4 acc = make_float4(0.f, 0.f, 0.f, 0.f);
#pragma unroll 8
  for (int c = 0; c < 64; ++c) {
    float4 v = px[(size_t)c * 16384u];                 // stride = one channel plane
    acc.x += v.x; acc.y += v.y; acc.z += v.z; acc.w += v.w;
  }
  const float s = 1.0f / 64.0f;
  reinterpret_cast<float4*>(xmean)[tid] = make_float4(acc.x * s, acc.y * s, acc.z * s, acc.w * s);
}

// ---------------------------------------------------------------------------
// Kernel 2: per-patch DCT projection + gate, via v_wmma_f32_16x16x32_f16.
// One wave32 handles 16 consecutive patches (M dim). GEMM1: (16x64)x(64x8->16)
// with two accumulating WMMAs (K=64). GEMM2: (16x8 padded to 32)x(8x64) with
// four WMMAs (N tiled by 16). Sigmoid on the f32 accumulators -> gate buffer.
// ---------------------------------------------------------------------------
__global__ __launch_bounds__(256) void k_dct_gate(const float* __restrict__ xmean,
                                                  const float* __restrict__ Wg,   // [64][8]
                                                  float* __restrict__ gate) {     // [16][64][32][32]
  __shared__ float lds_coeff[8][16][8];                // per-wave coeff staging
  const int lane    = threadIdx.x & 31;
  const int wave    = threadIdx.x >> 5;
  const int gwave   = blockIdx.x * 8 + wave;           // 0..1023
  const int baseP   = gwave << 4;                      // 16 patches per wave
  const int laneM   = lane & 15;
  const bool hiHalf = (lane >= 16);

  // patch coordinates for the A-matrix row this lane supplies (M = laneM)
  const int P  = baseP + laneM;                        // global patch id
  const int b  = P >> 10;
  const int hp = (P >> 5) & 31;
  const int wp = P & 31;

  // ---- A matrices (patch pixels, f16 16x32 layout; K = p*8+q) -------------
  // lanes 0-15: local K 0-7 ->(rows 0/4), 16-23 ->(rows 2/6); lanes 16-31: rows 1/5 and 3/7
  const float* xm = xmean + ((size_t)b << 16) + ((size_t)(hp << 3) << 8) + (wp << 3);
  const int par = hiHalf ? 1 : 0;
  v16h a0, a1;
  {
    const float* r0 = xm + (size_t)(par)     * 256;
    const float* r2 = xm + (size_t)(par + 2) * 256;
    const float* r4 = xm + (size_t)(par + 4) * 256;
    const float* r6 = xm + (size_t)(par + 6) * 256;
#pragma unroll
    for (int q = 0; q < 8; ++q) {
      a0[q]     = (_Float16)r0[q];   // local K = q        (global K or K+8)
      a0[8 + q] = (_Float16)r2[q];   // local K = 16 + q
      a1[q]     = (_Float16)r4[q];   // second WMMA, global K-32 in {0..7}
      a1[8 + q] = (_Float16)r6[q];
    }
  }

  // ---- B matrices (DCT bases, f16 32x16 layout; N = basis, zero-pad N>=8) --
  const int N = laneM;
  const unsigned u = (0x12201100u >> (4 * N)) & 0xFu;  // _MAP_X packed nibbles
  const unsigned v = (0x21021010u >> (4 * N)) & 0xFu;  // _MAP_Y packed nibbles
  const float valid = (N < 8) ? 1.0f : 0.0f;
  const float su = ((u != 0) ? 0.5f : 0.35355339059327373f) * valid; // sqrt2/sqrt8 or 1/sqrt8
  const float sv =  (v != 0) ? 0.5f : 0.35355339059327373f;
  const float uf = (float)u, vf = (float)v;
  const float p0  = hiHalf ? 1.0f : 0.0f;              // patch rows for WMMA1
  const float buA = dct1(p0,       uf, su);
  const float buB = dct1(p0 + 2.f, uf, su);
  const float buC = dct1(p0 + 4.f, uf, su);            // rows for WMMA2
  const float buD = dct1(p0 + 6.f, uf, su);
  float bvq[8];
#pragma unroll
  for (int q = 0; q < 8; ++q) bvq[q] = dct1((float)q, vf, sv);

  v16h b0, b1;
#pragma unroll
  for (int q = 0; q < 8; ++q) {
    b0[q]     = (_Float16)(buA * bvq[q]);
    b0[8 + q] = (_Float16)(buB * bvq[q]);
    b1[q]     = (_Float16)(buC * bvq[q]);
    b1[8 + q] = (_Float16)(buD * bvq[q]);
  }

  // ---- GEMM1: coeff = patches x bases^T  (two accumulating WMMAs, K=64) ----
  v8f c = {};
  c = __builtin_amdgcn_wmma_f32_16x16x32_f16(false, a0, false, b0, (short)0, c, false, false);
  c = __builtin_amdgcn_wmma_f32_16x16x32_f16(false, a1, false, b1, (short)0, c, false, false);

  // D layout: VGPR r -> (M=r, N=lane) for lanes 0-15, (M=8+r, N=lane-16) else
  if (N < 8) {
    const int m0 = hiHalf ? 8 : 0;
#pragma unroll
    for (int r = 0; r < 8; ++r) lds_coeff[wave][m0 + r][N] = c[r];
  }
  __syncthreads();

  // ---- GEMM2: pregate = coeff x W^T  (K=8 padded to 32, N=64 in 4 tiles) ---
  v16h a2 = {};
  if (!hiHalf) {
#pragma unroll
    for (int m = 0; m < 8; ++m) a2[m] = (_Float16)lds_coeff[wave][laneM][m];
  }

#pragma unroll
  for (int t = 0; t < 4; ++t) {
    v16h b2 = {};
    if (!hiHalf) {                                     // rows K=0..7 hold W[c][m]
      const float* wr = Wg + (size_t)((t << 4) + laneM) * 8;
#pragma unroll
      for (int m = 0; m < 8; ++m) b2[m] = (_Float16)wr[m];
    }
    v8f d = {};
    d = __builtin_amdgcn_wmma_f32_16x16x32_f16(false, a2, false, b2, (short)0, d, false, false);

    const int ch = (t << 4) + laneM;                   // channel = N + 16*t
    const int m0 = hiHalf ? 8 : 0;
#pragma unroll
    for (int r = 0; r < 8; ++r) {
      const int Pm  = baseP + m0 + r;
      const int bb  = Pm >> 10;
      const int hpp = (Pm >> 5) & 31;
      const int wpp = Pm & 31;
      const float gv = 1.0f / (1.0f + __expf(-d[r]));
      gate[((((size_t)bb << 6) + ch) << 10) + (hpp << 5) + wpp] = gv;
    }
  }
}

// ---------------------------------------------------------------------------
// Kernel 3: out = x * gate[b,c,h/8,w/8]   (streams 536 MB, float4 path)
// ---------------------------------------------------------------------------
__global__ __launch_bounds__(256) void k_apply(const float* __restrict__ x,
                                               const float* __restrict__ gate,
                                               float* __restrict__ out) {
  const size_t i4   = (size_t)blockIdx.x * 256 + threadIdx.x;  // 16,777,216 float4s
  const int    w4   = (int)(i4 & 63);                          // float4 column
  const size_t rest = i4 >> 6;
  const int    h    = (int)(rest & 255);
  const size_t bc   = rest >> 8;                               // b*64 + c, fused
  const float  g    = gate[(bc << 10) + (size_t)(((h >> 3) << 5) + (w4 >> 1))];
  float4 xv = reinterpret_cast<const float4*>(x)[i4];
  reinterpret_cast<float4*>(out)[i4] = make_float4(xv.x * g, xv.y * g, xv.z * g, xv.w * g);
}

// ---------------------------------------------------------------------------
extern "C" void kernel_launch(void* const* d_in, const int* in_sizes, int n_in,
                              void* d_out, int out_size, void* d_ws, size_t ws_size,
                              hipStream_t stream) {
  const float* x  = (const float*)d_in[0];   // (16, 64, 256, 256) f32
  const float* W  = (const float*)d_in[1];   // (64, 8) f32
  float* out      = (float*)d_out;           // (16, 64, 256, 256) f32

  float* xmean = (float*)d_ws;               // 16*256*256 f32  = 4 MB
  float* gate  = xmean + 16 * 256 * 256;     // 16*64*32*32 f32 = 4 MB

  (void)in_sizes; (void)n_in; (void)out_size; (void)ws_size;

  k_chan_mean<<<1024,  256, 0, stream>>>(x, xmean);
  k_dct_gate <<<128,   256, 0, stream>>>(xmean, W, gate);
  k_apply    <<<65536, 256, 0, stream>>>(x, gate, out);
}
